// CustomBayesianNeuralNetwork_38800734552074
// MI455X (gfx1250) — matrix operand order repaired and checked by reading, not compiled, not run on hardware
//
#include <hip/hip_runtime.h>
#include <hip/hip_bf16.h>

// ---------------------------------------------------------------------------
// Bayesian LSTM (MC sampling) for MI455X / gfx1250.
// Per-wave 16-row batch tile, v_wmma_f32_16x16x32_f16 for all matmuls,
// v_rcp/v_exp TRANS-unit activations that co-execute with the WMMA pipe.
// ---------------------------------------------------------------------------

typedef __attribute__((ext_vector_type(16))) _Float16 v16h;
typedef __attribute__((ext_vector_type(8)))  float    v8f;

union V16H { v16h v; _Float16 h[16]; };

#define TSEQ   30
#define FIN    14
#define HID    32
#define GATES  128   // 4*HID
#define LMC    10

__device__ __forceinline__ float sp_f(float x) {        // softplus (fast)
    return __logf(1.0f + __expf(x));
}
__device__ __forceinline__ float sig_f(float x) {       // sigmoid via v_rcp_f32
    return __builtin_amdgcn_rcpf(1.0f + __expf(-x));
}
__device__ __forceinline__ float tanh_f(float x) {      // tanh via v_exp + v_rcp
    float e = __expf(-2.0f * x);
    return (1.0f - e) * __builtin_amdgcn_rcpf(1.0f + e);
}

// In-wave LDS ordering: per ISA, a wave's DS ops complete in order (DScnt);
// wait for them and stop the compiler from caching LDS values across this point.
__device__ __forceinline__ void wave_lds_fence() {
    asm volatile("s_wait_dscnt 0" ::: "memory");
}

__global__ __launch_bounds__(128) void
bnn_lstm_wmma_kernel(const float* __restrict__ x,
                     const float* __restrict__ mu_ih, const float* __restrict__ rho_ih,
                     const float* __restrict__ mu_hh, const float* __restrict__ rho_hh,
                     const float* __restrict__ mu_b,  const float* __restrict__ rho_b,
                     const float* __restrict__ l1_mu_w, const float* __restrict__ l1_rho_w,
                     const float* __restrict__ l1_mu_b, const float* __restrict__ l1_rho_b,
                     const float* __restrict__ l2_mu_w, const float* __restrict__ l2_rho_w,
                     const float* __restrict__ l2_mu_b, const float* __restrict__ l2_rho_b,
                     const float* __restrict__ eps_ih, const float* __restrict__ eps_hh,
                     const float* __restrict__ eps_b,
                     const float* __restrict__ eps1_w, const float* __restrict__ eps1_b,
                     const float* __restrict__ eps2_w, const float* __restrict__ eps2_b,
                     float* __restrict__ out)
{
    // Combined [w_ih | w_hh] as B-fragments: (ntile 0..7, kstep 0..1, lane 0..31) x 16 halfs.
    __shared__ __align__(32) _Float16 Wfrag[8 * 2 * 32 * 16];   // 16 KB
    __shared__ __align__(16) float    blds[GATES];              // sampled LSTM bias
    __shared__ __align__(32) _Float16 w1f[32 * 16];             // layer-1 B fragments (1 KB)
    __shared__ __align__(16) float    b1s[16];
    __shared__ __align__(16) float    w2s[16];
    __shared__ float                  b2s;
    __shared__ __align__(32) _Float16 hlds[4][16 * HID];        // per-wave hidden state (4 KB)
    __shared__ __align__(16) float    out1[4][16 * 16];         // per-wave layer-1 output (4 KB)

    const int tid  = threadIdx.x;
    const int wave = tid >> 5;
    const int lane = tid & 31;
    const int hv   = lane >> 4;        // lane-half select
    const int n    = lane & 15;        // B/C/D column owned by this lane
    const int m    = n;                // A row owned by this lane
    const size_t brow = (size_t)blockIdx.x * 64 + (size_t)wave * 16;

    float acc_out = 0.0f;              // lanes 0..15: running MC sum for row brow+lane

    for (int s = 0; s < LMC; ++s) {
        // ---------------- sample weights into LDS (block-cooperative) ------
        {   // zero-fill fragment store (covers K padding 14,15 and 48..63)
            int* wz = (int*)Wfrag;
            #pragma unroll 4
            for (int i = tid; i < 4096; i += 128) wz[i] = 0;
        }
        __syncthreads();

        // w_ih -> K 0..13
        for (int i = tid; i < GATES * FIN; i += 128) {
            int g = i / FIN, k = i % FIN;
            float w = mu_ih[i] + sp_f(rho_ih[i]) * eps_ih[s * GATES * FIN + i];
            int nt = g >> 4, nn = g & 15;
            int ln = (k < 16) ? nn : 16 + nn;                 // k<14 always
            Wfrag[((nt * 2 + 0) * 32 + ln) * 16 + (k & 15)] = (_Float16)w;
        }
        // w_hh -> K 16..47
        for (int i = tid; i < GATES * HID; i += 128) {
            int g = i / HID, kg = (i % HID) + 16;
            float w = mu_hh[i] + sp_f(rho_hh[i]) * eps_hh[s * GATES * HID + i];
            int nt = g >> 4, nn = g & 15;
            int ks = kg >> 5, kl = kg & 31;
            int ln = (kl < 16) ? nn : 16 + nn;
            Wfrag[((nt * 2 + ks) * 32 + ln) * 16 + (kl & 15)] = (_Float16)w;
        }
        if (tid < GATES)
            blds[tid] = mu_b[tid] + sp_f(rho_b[tid]) * eps_b[s * GATES + tid];
        // layer-1 weights (16 x 32) -> B fragments
        for (int i = tid; i < 16 * HID; i += 128) {
            int o = i >> 5, k = i & 31;
            float w = l1_mu_w[i] + sp_f(l1_rho_w[i]) * eps1_w[s * 16 * HID + i];
            int ln = (k < 16) ? o : 16 + o;
            w1f[ln * 16 + (k & 15)] = (_Float16)w;
        }
        if (tid < 16) b1s[tid] = l1_mu_b[tid] + sp_f(l1_rho_b[tid]) * eps1_b[s * 16 + tid];
        if (tid < 16) w2s[tid] = l2_mu_w[tid] + sp_f(l2_rho_w[tid]) * eps2_w[s * 16 + tid];
        if (tid == 0) b2s = l2_mu_b[0] + sp_f(l2_rho_b[0]) * eps2_b[s];

        {   // zero hidden state for this wave (16*32 halfs / 32 lanes = 16 halfs each)
            V16H z;
            #pragma unroll
            for (int j = 0; j < 16; ++j) z.h[j] = (_Float16)0.0f;
            *(v16h*)(&hlds[wave][lane * 16]) = z.v;
        }
        __syncthreads();

        float c[2][8];                 // cell state: this lane's (j = t2*16+n, row = hv*8+r)
        #pragma unroll
        for (int t2 = 0; t2 < 2; ++t2)
            #pragma unroll
            for (int r = 0; r < 8; ++r) c[t2][r] = 0.0f;

        const float* xr0 = x + (brow + (size_t)m) * (TSEQ * FIN);

        // ------------------------- recurrence ------------------------------
        for (int t = 0; t < TSEQ; ++t) {
            const float* xr = xr0 + t * FIN;
            __builtin_prefetch(xr + FIN, 0, 3);   // speculative: OOB drops silently

            // A fragments: K 0..15 = padded x_t, K 16..47 = h, K 48..63 = 0
            V16H a0, a1;
            const _Float16* hl = &hlds[wave][m * HID];
            #pragma unroll
            for (int j = 0; j < 16; ++j) {
                a0.h[j] = hv ? hl[j] : ((j < FIN) ? (_Float16)xr[j] : (_Float16)0.0f);   // K0..15 | K16..31
                a1.h[j] = hv ? (_Float16)0.0f : hl[16 + j];                              // K32..47 | K48..63
            }

            v8f acc[8];
            #pragma unroll
            for (int nt = 0; nt < 8; ++nt) {
                float bg = blds[nt * 16 + n];            // bias, free via C operand
                #pragma unroll
                for (int r = 0; r < 8; ++r) acc[nt][r] = bg;
                v16h b0 = *(const v16h*)(&Wfrag[((nt * 2 + 0) * 32 + lane) * 16]);
                acc[nt] = __builtin_amdgcn_wmma_f32_16x16x32_f16(
                              false, a0.v, false, b0, (short)0, acc[nt], false, false);
                v16h b1v = *(const v16h*)(&Wfrag[((nt * 2 + 1) * 32 + lane) * 16]);
                acc[nt] = __builtin_amdgcn_wmma_f32_16x16x32_f16(
                              false, a1.v, false, b1v, (short)0, acc[nt], false, false);
            }

            // pointwise LSTM update; gate tiles: [0,1]=i [2,3]=f [4,5]=g [6,7]=o
            #pragma unroll
            for (int t2 = 0; t2 < 2; ++t2) {
                #pragma unroll
                for (int r = 0; r < 8; ++r) {
                    float ig = sig_f(acc[0 + t2][r]);
                    float fg = sig_f(acc[2 + t2][r]);
                    float gg = tanh_f(acc[4 + t2][r]);
                    float og = sig_f(acc[6 + t2][r]);
                    float cn = fg * c[t2][r] + ig * gg;
                    c[t2][r] = cn;
                    float hn = og * tanh_f(cn);
                    int mrow = hv * 8 + r;
                    hlds[wave][mrow * HID + t2 * 16 + n] = (_Float16)hn;
                }
            }
            // per-wave LDS state; in-wave DS ordering is enough (no block barrier)
            wave_lds_fence();
        }

        // ------------------------- head: h @ w1^T + b1 ---------------------
        {
            V16H ah;
            const _Float16* hl = &hlds[wave][m * HID];
            #pragma unroll
            for (int j = 0; j < 16; ++j)
                ah.h[j] = hl[hv * 16 + j];
            v8f oa;
            float bg = b1s[n];
            #pragma unroll
            for (int r = 0; r < 8; ++r) oa[r] = bg;
            v16h bO = *(const v16h*)(&w1f[lane * 16]);
            oa = __builtin_amdgcn_wmma_f32_16x16x32_f16(
                     false, ah.v, false, bO, (short)0, oa, false, false);
            #pragma unroll
            for (int r = 0; r < 8; ++r)
                out1[wave][(hv * 8 + r) * 16 + n] = oa[r];
        }
        wave_lds_fence();

        // ------------------------- head: out1 @ w2^T + b2 ------------------
        if (lane < 16) {
            float s2 = b2s;
            #pragma unroll
            for (int j = 0; j < 16; ++j)
                s2 += out1[wave][lane * 16 + j] * w2s[j];
            acc_out += s2;
        }
        __syncthreads();   // protect shared Wfrag/blds before next sample refill
    }

    if (lane < 16)
        out[brow + (size_t)lane] = acc_out * (1.0f / (float)LMC);
}

extern "C" void kernel_launch(void* const* d_in, const int* in_sizes, int n_in,
                              void* d_out, int out_size, void* d_ws, size_t ws_size,
                              hipStream_t stream) {
    (void)n_in; (void)d_ws; (void)ws_size; (void)out_size;

    const float* x        = (const float*)d_in[0];
    const float* mu_ih    = (const float*)d_in[1];
    const float* rho_ih   = (const float*)d_in[2];
    const float* mu_hh    = (const float*)d_in[3];
    const float* rho_hh   = (const float*)d_in[4];
    const float* mu_b     = (const float*)d_in[5];
    const float* rho_b    = (const float*)d_in[6];
    const float* l1_mu_w  = (const float*)d_in[7];
    const float* l1_rho_w = (const float*)d_in[8];
    const float* l1_mu_b  = (const float*)d_in[9];
    const float* l1_rho_b = (const float*)d_in[10];
    const float* l2_mu_w  = (const float*)d_in[11];
    const float* l2_rho_w = (const float*)d_in[12];
    const float* l2_mu_b  = (const float*)d_in[13];
    const float* l2_rho_b = (const float*)d_in[14];
    const float* eps_ih   = (const float*)d_in[15];
    const float* eps_hh   = (const float*)d_in[16];
    const float* eps_b    = (const float*)d_in[17];
    const float* eps1_w   = (const float*)d_in[18];
    const float* eps1_b   = (const float*)d_in[19];
    const float* eps2_w   = (const float*)d_in[20];
    const float* eps2_b   = (const float*)d_in[21];
    float*       out      = (float*)d_out;

    const int Btot   = in_sizes[0] / (TSEQ * FIN);   // 32768
    const int blocks = (Btot + 63) / 64;             // 64 batch rows per block

    bnn_lstm_wmma_kernel<<<blocks, 128, 0, stream>>>(
        x, mu_ih, rho_ih, mu_hh, rho_hh, mu_b, rho_b,
        l1_mu_w, l1_rho_w, l1_mu_b, l1_rho_b,
        l2_mu_w, l2_rho_w, l2_mu_b, l2_rho_b,
        eps_ih, eps_hh, eps_b, eps1_w, eps1_b, eps2_w, eps2_b,
        out);
}
